// TransformerBlock_22204980920451
// MI455X (gfx1250) — matrix operand order repaired and checked by reading, not compiled, not run on hardware
//
#include <hip/hip_runtime.h>

// ---------------------------------------------------------------------------
// Types / helpers
// ---------------------------------------------------------------------------
typedef __attribute__((ext_vector_type(16))) __bf16 v16bf;
typedef __attribute__((ext_vector_type(8)))  float  v8f;
typedef int v4i __attribute__((vector_size(16)));

union AFrag { v16bf v; unsigned int u[8]; };

__device__ __forceinline__ unsigned short f2bf(float f) {
  unsigned int u = __float_as_uint(f);
  unsigned int r = (u + 0x7FFFu + ((u >> 16) & 1u)) >> 16;
  return (unsigned short)r;
}
__device__ __forceinline__ float bf2f(unsigned short h) {
  return __uint_as_float(((unsigned int)h) << 16);
}

#define NB 2
#define SEQ 2048
#define DM 2048
#define MROWS 4096      // B*L
#define KVW 512         // N_GROUPS * D_HEAD
#define HDIM 5632
#define NHEADS 16

// ---------------------------------------------------------------------------
// CDNA5 async global->LDS copy (no VGPR round trip), with sync fallback
// ---------------------------------------------------------------------------
#if defined(__has_builtin)
#if __has_builtin(__builtin_amdgcn_global_load_async_to_lds_b128)
#define USE_ASYNC_LDS 1
#endif
#endif

__device__ __forceinline__ void cp16_to_lds(unsigned short* lds,
                                            const unsigned short* g) {
#ifdef USE_ASYNC_LDS
  __builtin_amdgcn_global_load_async_to_lds_b128(
      (__attribute__((address_space(1))) v4i*)(g),
      (__attribute__((address_space(3))) v4i*)(lds), 0, 0);
#else
  *(uint4*)lds = *(const uint4*)g;
#endif
}

__device__ __forceinline__ void async_wait_all() {
#ifdef USE_ASYNC_LDS
#if __has_builtin(__builtin_amdgcn_s_wait_asynccnt)
  __builtin_amdgcn_s_wait_asynccnt(0);
#else
  asm volatile("s_wait_asynccnt 0" ::: "memory");
#endif
#endif
}

// ---------------------------------------------------------------------------
// fp32 -> bf16 conversion (weights)
// ---------------------------------------------------------------------------
__global__ void cvt_kernel(const float* __restrict__ in,
                           unsigned short* __restrict__ out, size_t n) {
  size_t i = (size_t)blockIdx.x * blockDim.x + threadIdx.x;
  size_t stride = (size_t)gridDim.x * blockDim.x;
  for (; i < n; i += stride) out[i] = f2bf(in[i]);
}

// ---------------------------------------------------------------------------
// RMSNorm: one block per row of D=2048, fp32 in -> bf16 out
// ---------------------------------------------------------------------------
__global__ __launch_bounds__(256) void rmsnorm_kernel(
    const float* __restrict__ x, const float* __restrict__ g,
    unsigned short* __restrict__ out) {
  __shared__ float red[8];
  int row = blockIdx.x;
  const float* xr = x + (size_t)row * DM;
  float s = 0.f;
  for (int c = threadIdx.x; c < DM; c += 256) { float v = xr[c]; s += v * v; }
  for (int off = 1; off < 32; off <<= 1) s += __shfl_xor(s, off, 32);
  if ((threadIdx.x & 31) == 0) red[threadIdx.x >> 5] = s;
  __syncthreads();
  float tot = 0.f;
#pragma unroll
  for (int i = 0; i < 8; ++i) tot += red[i];
  float inv = rsqrtf(tot / (float)DM + 1e-6f);
  for (int c = threadIdx.x; c < DM; c += 256)
    out[(size_t)row * DM + c] = f2bf(xr[c] * inv * g[c]);
}

// ---------------------------------------------------------------------------
// Generic bf16 WMMA GEMM: C[M,N] = A[M,K] @ W[K,N] + bias (+ residual)
// Block: 256 thr (8 waves). Block tile 256x64. Wave: 32 rows x 64 cols,
// 8x v_wmma_f32_16x16x32_bf16 per K-step of 32. A staged via async->LDS.
// ---------------------------------------------------------------------------
template <bool OUT_BF16, bool ADD_RES>
__global__ __launch_bounds__(256) void gemm_bf16(
    const unsigned short* __restrict__ A, const unsigned short* __restrict__ W,
    const float* __restrict__ bias, const float* __restrict__ res,
    void* __restrict__ outp, int M, int N, int K) {
  __shared__ unsigned short As[256 * 40];  // padded stride 40
  __shared__ unsigned short Bs[64 * 36];   // transposed [n][k], stride 36
  int tid = threadIdx.x, lane = tid & 31, w = tid >> 5;
  int l15 = lane & 15, hh = lane >> 4;
  int m0 = blockIdx.x * 256, n0 = blockIdx.y * 64;

  v8f acc[2][4];
#pragma unroll
  for (int s = 0; s < 2; ++s)
#pragma unroll
    for (int t = 0; t < 4; ++t)
#pragma unroll
      for (int j = 0; j < 8; ++j) acc[s][t][j] = 0.f;

  for (int k0 = 0; k0 < K; k0 += 32) {
    // stage A tile 256x32 (async global->LDS, 4 x b128 per thread)
#pragma unroll
    for (int it = 0; it < 4; ++it) {
      int idx = tid + it * 256;            // 1024 vectors of 8 bf16
      int r = idx >> 2, c = (idx & 3) * 8;
      cp16_to_lds(&As[r * 40 + c], A + (size_t)(m0 + r) * K + k0 + c);
    }
    // stage W tile 32x64, transposed into Bs[n][k]
    {
      int r = tid >> 3, c = (tid & 7) * 8;
      uint4 d = *(const uint4*)(W + (size_t)(k0 + r) * N + n0 + c);
      const unsigned short* pd = (const unsigned short*)&d;
#pragma unroll
      for (int j = 0; j < 8; ++j) Bs[(c + j) * 36 + r] = pd[j];
    }
    async_wait_all();
    __syncthreads();

    AFrag a0, a1;
    int am = w * 32 + l15;
#pragma unroll
    for (int i = 0; i < 8; ++i) {
      int kk = 2 * i + 16 * hh;
      a0.u[i] = *(const unsigned int*)&As[am * 40 + kk];
      a1.u[i] = *(const unsigned int*)&As[(am + 16) * 40 + kk];
    }
#pragma unroll
    for (int t = 0; t < 4; ++t) {
      AFrag b;
      int bn = t * 16 + l15;
#pragma unroll
      for (int i = 0; i < 8; ++i) {
        int kk = 2 * i + 16 * hh;
        b.u[i] = *(const unsigned int*)&Bs[bn * 36 + kk];
      }
      acc[0][t] = __builtin_amdgcn_wmma_f32_16x16x32_bf16(
          false, a0.v, false, b.v, (short)0, acc[0][t], false, false);
      acc[1][t] = __builtin_amdgcn_wmma_f32_16x16x32_bf16(
          false, a1.v, false, b.v, (short)0, acc[1][t], false, false);
    }
    __syncthreads();
  }

#pragma unroll
  for (int s = 0; s < 2; ++s)
#pragma unroll
    for (int t = 0; t < 4; ++t)
#pragma unroll
      for (int i = 0; i < 8; ++i) {
        int row = m0 + w * 32 + s * 16 + i + 8 * hh;
        int col = n0 + t * 16 + l15;
        float vv = acc[s][t][i] + bias[col];
        if (ADD_RES) vv += res[(size_t)row * N + col];
        if (OUT_BF16)
          ((unsigned short*)outp)[(size_t)row * N + col] = f2bf(vv);
        else
          ((float*)outp)[(size_t)row * N + col] = vv;
      }
}

// ---------------------------------------------------------------------------
// Fused SwiGLU GEMM: out = bf16( silu(A@W1 + b1) * (A@Wg + bg) )
// Block tile 128x64, wave 16x64, two accumulator sets sharing the A fragment.
// ---------------------------------------------------------------------------
__global__ __launch_bounds__(256) void gemm_swiglu(
    const unsigned short* __restrict__ A, const unsigned short* __restrict__ W1,
    const unsigned short* __restrict__ Wg, const float* __restrict__ b1,
    const float* __restrict__ bg, unsigned short* __restrict__ outp, int M,
    int N, int K) {
  __shared__ unsigned short As[128 * 40];
  __shared__ unsigned short B1s[64 * 36];
  __shared__ unsigned short Bgs[64 * 36];
  int tid = threadIdx.x, lane = tid & 31, w = tid >> 5;
  int l15 = lane & 15, hh = lane >> 4;
  int m0 = blockIdx.x * 128, n0 = blockIdx.y * 64;

  v8f acc1[4], accg[4];
#pragma unroll
  for (int t = 0; t < 4; ++t)
#pragma unroll
    for (int j = 0; j < 8; ++j) { acc1[t][j] = 0.f; accg[t][j] = 0.f; }

  for (int k0 = 0; k0 < K; k0 += 32) {
#pragma unroll
    for (int it = 0; it < 2; ++it) {
      int idx = tid + it * 256;            // 512 vectors of 8 bf16
      int r = idx >> 2, c = (idx & 3) * 8;
      cp16_to_lds(&As[r * 40 + c], A + (size_t)(m0 + r) * K + k0 + c);
    }
    {
      int r = tid >> 3, c = (tid & 7) * 8;
      size_t gi = (size_t)(k0 + r) * N + n0 + c;
      uint4 d1 = *(const uint4*)(W1 + gi);
      uint4 dg = *(const uint4*)(Wg + gi);
      const unsigned short* p1 = (const unsigned short*)&d1;
      const unsigned short* pg = (const unsigned short*)&dg;
#pragma unroll
      for (int j = 0; j < 8; ++j) {
        B1s[(c + j) * 36 + r] = p1[j];
        Bgs[(c + j) * 36 + r] = pg[j];
      }
    }
    async_wait_all();
    __syncthreads();

    AFrag a;
    int am = w * 16 + l15;
#pragma unroll
    for (int i = 0; i < 8; ++i) {
      int kk = 2 * i + 16 * hh;
      a.u[i] = *(const unsigned int*)&As[am * 40 + kk];
    }
#pragma unroll
    for (int t = 0; t < 4; ++t) {
      AFrag b1f, bgf;
      int bn = t * 16 + l15;
#pragma unroll
      for (int i = 0; i < 8; ++i) {
        int kk = 2 * i + 16 * hh;
        b1f.u[i] = *(const unsigned int*)&B1s[bn * 36 + kk];
        bgf.u[i] = *(const unsigned int*)&Bgs[bn * 36 + kk];
      }
      acc1[t] = __builtin_amdgcn_wmma_f32_16x16x32_bf16(
          false, a.v, false, b1f.v, (short)0, acc1[t], false, false);
      accg[t] = __builtin_amdgcn_wmma_f32_16x16x32_bf16(
          false, a.v, false, bgf.v, (short)0, accg[t], false, false);
    }
    __syncthreads();
  }

#pragma unroll
  for (int t = 0; t < 4; ++t)
#pragma unroll
    for (int i = 0; i < 8; ++i) {
      int row = m0 + w * 16 + i + 8 * hh;
      int col = n0 + t * 16 + l15;
      float v1 = acc1[t][i] + b1[col];
      float vg = accg[t][i] + bg[col];
      float si = v1 / (1.f + __expf(-v1));
      outp[(size_t)row * N + col] = f2bf(si * vg);
    }
}

// ---------------------------------------------------------------------------
// In-place RoPE on bf16 [B, L, Wd] (Wd = 2048 for q, 512 for k), head dim 128
// ---------------------------------------------------------------------------
__global__ void rope_kernel(unsigned short* __restrict__ qk, int Wd) {
  size_t idx = (size_t)blockIdx.x * blockDim.x + threadIdx.x;  // pair index
  int ppr = Wd / 2;
  size_t row = idx / ppr;
  int p = (int)(idx % ppr);
  int l = (int)(row % SEQ);
  int hd = p & 63;  // pair index within a 128-dim head
  float inv = __powf(10000.f, -(float)(2 * hd) / 128.f);
  float ang = (float)l * inv;
  float sn, cs;
  __sincosf(ang, &sn, &cs);
  unsigned short* ptr = qk + row * (size_t)Wd + 2 * p;
  float xe = bf2f(ptr[0]), xo = bf2f(ptr[1]);
  ptr[0] = f2bf(xe * cs - xo * sn);
  ptr[1] = f2bf(xe * sn + xo * cs);
}

// ---------------------------------------------------------------------------
// Flash attention, causal GQA. One wave (32 thr) per (16 q-rows, head, batch).
// Q/K/V bf16 post-RoPE; K^T fragments read directly from global (d contiguous),
// V staged transposed into LDS with key-pairs packed per dword.
// ---------------------------------------------------------------------------
__global__ __launch_bounds__(32) void attn_kernel(
    const unsigned short* __restrict__ Q,   // [B, L, 2048]
    const unsigned short* __restrict__ Kc,  // [B, L, 512]
    const unsigned short* __restrict__ V,   // [B, L, 512]
    unsigned short* __restrict__ O) {       // [B, L, 2048]
  __shared__ unsigned short Ps[16 * 32];
  __shared__ unsigned int Vt[128 * 17];  // [d][keypair] packed dwords
  int lane = threadIdx.x, l15 = lane & 15, hh = lane >> 4;
  int qb = blockIdx.x * 16;
  int h = blockIdx.y, bi = blockIdx.z;
  int g = h >> 2;  // 4 heads per group
  const float scale = 0.08838834764831845f;  // 1/sqrt(128)

  AFrag qa[4];
  const unsigned short* Qp =
      Q + ((size_t)(bi * SEQ + qb + l15)) * DM + h * 128;
#pragma unroll
  for (int c = 0; c < 4; ++c)
#pragma unroll
    for (int i = 0; i < 8; ++i) {
      int kk = 2 * i + 16 * hh + 32 * c;
      qa[c].u[i] = *(const unsigned int*)(Qp + kk);
    }

  v8f o[8];
  float rmax[8], rsum[8];
#pragma unroll
  for (int t = 0; t < 8; ++t)
#pragma unroll
    for (int j = 0; j < 8; ++j) o[t][j] = 0.f;
#pragma unroll
  for (int i = 0; i < 8; ++i) { rmax[i] = -1e30f; rsum[i] = 0.f; }

  for (int kb = 0; kb < qb + 16; kb += 32) {
    // ---- stage V^T: 32 keys x 128 d, packed key-pairs per dword ----
    {
      int p = l15;        // key pair 0..15
      int dh = hh * 64;   // d half
      const unsigned short* v0 =
          V + ((size_t)(bi * SEQ + kb + 2 * p)) * KVW + g * 128 + dh;
      const unsigned short* v1 = v0 + KVW;
#pragma unroll
      for (int j = 0; j < 8; ++j) {
        uint4 a = *(const uint4*)(v0 + j * 8);
        uint4 b = *(const uint4*)(v1 + j * 8);
        const unsigned short* pa = (const unsigned short*)&a;
        const unsigned short* pb = (const unsigned short*)&b;
#pragma unroll
        for (int e = 0; e < 8; ++e)
          Vt[(dh + j * 8 + e) * 17 + p] =
              (unsigned int)pa[e] | ((unsigned int)pb[e] << 16);
      }
    }
    __builtin_amdgcn_wave_barrier();

    // ---- S = Q @ K^T for 32 keys (two 16-wide tiles) ----
    v8f s0, s1;
#pragma unroll
    for (int j = 0; j < 8; ++j) { s0[j] = 0.f; s1[j] = 0.f; }
#pragma unroll
    for (int c = 0; c < 4; ++c) {
      AFrag kf0, kf1;
#pragma unroll
      for (int i = 0; i < 8; ++i) {
        int kd = 32 * c + 2 * i + 16 * hh;
        size_t base =
            ((size_t)(bi * SEQ + kb + l15)) * KVW + g * 128 + kd;
        kf0.u[i] = *(const unsigned int*)(Kc + base);
        kf1.u[i] = *(const unsigned int*)(Kc + base + (size_t)16 * KVW);
      }
      s0 = __builtin_amdgcn_wmma_f32_16x16x32_bf16(false, qa[c].v, false,
                                                   kf0.v, (short)0, s0,
                                                   false, false);
      s1 = __builtin_amdgcn_wmma_f32_16x16x32_bf16(false, qa[c].v, false,
                                                   kf1.v, (short)0, s1,
                                                   false, false);
    }

    // ---- causal mask + online softmax update ----
#pragma unroll
    for (int i = 0; i < 8; ++i) {
      int m = i + 8 * hh;
      int qrow = qb + m;
      float a0 = (kb + l15 <= qrow) ? s0[i] * scale : -1e30f;
      float a1 = (kb + 16 + l15 <= qrow) ? s1[i] * scale : -1e30f;
      float mx = fmaxf(a0, a1);
#pragma unroll
      for (int off = 1; off < 16; off <<= 1)
        mx = fmaxf(mx, __shfl_xor(mx, off, 32));
      float nm = fmaxf(rmax[i], mx);
      float cf = __expf(rmax[i] - nm);
      float p0 = __expf(a0 - nm);
      float p1 = __expf(a1 - nm);
      float ss = p0 + p1;
#pragma unroll
      for (int off = 1; off < 16; off <<= 1) ss += __shfl_xor(ss, off, 32);
      rsum[i] = rsum[i] * cf + ss;
      rmax[i] = nm;
#pragma unroll
      for (int t = 0; t < 8; ++t) o[t][i] *= cf;
      Ps[m * 32 + l15] = f2bf(p0);
      Ps[m * 32 + 16 + l15] = f2bf(p1);
    }
    __builtin_amdgcn_wave_barrier();

    // ---- O += P @ V ----
    AFrag pf;
#pragma unroll
    for (int i = 0; i < 8; ++i) {
      int kk = 2 * i + 16 * hh;
      pf.u[i] = *(const unsigned int*)&Ps[l15 * 32 + kk];
    }
#pragma unroll
    for (int t = 0; t < 8; ++t) {
      AFrag vf;
      int nd = t * 16 + l15;
#pragma unroll
      for (int i = 0; i < 8; ++i) vf.u[i] = Vt[nd * 17 + (i + 8 * hh)];
      o[t] = __builtin_amdgcn_wmma_f32_16x16x32_bf16(
          false, pf.v, false, vf.v, (short)0, o[t], false, false);
    }
    __builtin_amdgcn_wave_barrier();
  }

  // ---- normalize + store ----
#pragma unroll
  for (int t = 0; t < 8; ++t)
#pragma unroll
    for (int i = 0; i < 8; ++i) {
      int m = i + 8 * hh;
      float vv = o[t][i] / rsum[i];
      O[((size_t)(bi * SEQ + qb + m)) * DM + h * 128 + t * 16 + l15] =
          f2bf(vv);
    }
}

// ---------------------------------------------------------------------------
// Host orchestration
// ---------------------------------------------------------------------------
extern "C" void kernel_launch(void* const* d_in, const int* in_sizes, int n_in,
                              void* d_out, int out_size, void* d_ws,
                              size_t ws_size, hipStream_t stream) {
  (void)in_sizes; (void)n_in; (void)out_size; (void)ws_size;
  const float* x  = (const float*)d_in[0];
  const float* g1 = (const float*)d_in[1];
  const float* wq = (const float*)d_in[2];
  const float* bq = (const float*)d_in[3];
  const float* wk = (const float*)d_in[4];
  const float* bk = (const float*)d_in[5];
  const float* wv = (const float*)d_in[6];
  const float* bv = (const float*)d_in[7];
  const float* wo = (const float*)d_in[8];
  const float* bo = (const float*)d_in[9];
  const float* g2 = (const float*)d_in[10];
  const float* w1 = (const float*)d_in[11];
  const float* b1 = (const float*)d_in[12];
  const float* wg = (const float*)d_in[13];
  const float* bg = (const float*)d_in[14];
  const float* w2 = (const float*)d_in[15];
  const float* b2 = (const float*)d_in[16];
  float* out = (float*)d_out;

  char* p = (char*)d_ws;
  auto alloc = [&](size_t bytes) -> char* {
    char* r = p;
    p += (bytes + 255) & ~(size_t)255;
    return r;
  };
  unsigned short* h_b    = (unsigned short*)alloc((size_t)MROWS * DM * 2);
  unsigned short* wq_b   = (unsigned short*)alloc((size_t)DM * DM * 2);
  unsigned short* wk_b   = (unsigned short*)alloc((size_t)DM * KVW * 2);
  unsigned short* wv_b   = (unsigned short*)alloc((size_t)DM * KVW * 2);
  unsigned short* wo_b   = (unsigned short*)alloc((size_t)DM * DM * 2);
  unsigned short* w1_b   = (unsigned short*)alloc((size_t)DM * HDIM * 2);
  unsigned short* wg_b   = (unsigned short*)alloc((size_t)DM * HDIM * 2);
  unsigned short* w2_b   = (unsigned short*)alloc((size_t)HDIM * DM * 2);
  unsigned short* q_b    = (unsigned short*)alloc((size_t)MROWS * DM * 2);
  unsigned short* k_b    = (unsigned short*)alloc((size_t)MROWS * KVW * 2);
  unsigned short* v_b    = (unsigned short*)alloc((size_t)MROWS * KVW * 2);
  unsigned short* attn_b = (unsigned short*)alloc((size_t)MROWS * DM * 2);
  float*          x2     = (float*)alloc((size_t)MROWS * DM * 4);
  unsigned short* a_b    = (unsigned short*)alloc((size_t)MROWS * HDIM * 2);

  // weights fp32 -> bf16 (hot in L2)
  cvt_kernel<<<2048, 256, 0, stream>>>(wq, wq_b, (size_t)DM * DM);
  cvt_kernel<<<2048, 256, 0, stream>>>(wk, wk_b, (size_t)DM * KVW);
  cvt_kernel<<<2048, 256, 0, stream>>>(wv, wv_b, (size_t)DM * KVW);
  cvt_kernel<<<2048, 256, 0, stream>>>(wo, wo_b, (size_t)DM * DM);
  cvt_kernel<<<2048, 256, 0, stream>>>(w1, w1_b, (size_t)DM * HDIM);
  cvt_kernel<<<2048, 256, 0, stream>>>(wg, wg_b, (size_t)DM * HDIM);
  cvt_kernel<<<2048, 256, 0, stream>>>(w2, w2_b, (size_t)HDIM * DM);

  // h = rmsnorm(x, g1)
  rmsnorm_kernel<<<MROWS, 256, 0, stream>>>(x, g1, h_b);

  // q,k,v projections
  gemm_bf16<true, false><<<dim3(MROWS / 256, DM / 64), 256, 0, stream>>>(
      h_b, wq_b, bq, nullptr, q_b, MROWS, DM, DM);
  gemm_bf16<true, false><<<dim3(MROWS / 256, KVW / 64), 256, 0, stream>>>(
      h_b, wk_b, bk, nullptr, k_b, MROWS, KVW, DM);
  gemm_bf16<true, false><<<dim3(MROWS / 256, KVW / 64), 256, 0, stream>>>(
      h_b, wv_b, bv, nullptr, v_b, MROWS, KVW, DM);

  // RoPE in place
  rope_kernel<<<(int)(((size_t)MROWS * DM / 2) / 256), 256, 0, stream>>>(q_b, DM);
  rope_kernel<<<(int)(((size_t)MROWS * KVW / 2) / 256), 256, 0, stream>>>(k_b, KVW);

  // attention
  attn_kernel<<<dim3(SEQ / 16, NHEADS, NB), 32, 0, stream>>>(q_b, k_b, v_b,
                                                             attn_b);

  // x2 = x + attn @ wo + bo
  gemm_bf16<false, true><<<dim3(MROWS / 256, DM / 64), 256, 0, stream>>>(
      attn_b, wo_b, bo, x, x2, MROWS, DM, DM);

  // h2 = rmsnorm(x2, g2) (reuse h_b)
  rmsnorm_kernel<<<MROWS, 256, 0, stream>>>(x2, g2, h_b);

  // fused FFN up+gate: a = silu(h2@w1+b1) * (h2@wg+bg)
  gemm_swiglu<<<dim3(MROWS / 128, HDIM / 64), 256, 0, stream>>>(
      h_b, w1_b, wg_b, b1, bg, a_b, MROWS, HDIM, DM);

  // out = x2 + a @ w2 + b2
  gemm_bf16<false, true><<<dim3(MROWS / 256, DM / 64), 256, 0, stream>>>(
      a_b, w2_b, b2, x2, out, MROWS, DM, HDIM);
}
